// LatentDirectionEncoder_38878043963447
// MI455X (gfx1250) — compile-verified, matrix operand order checked by reading, not run on hardware
//
#include <hip/hip_runtime.h>
#include <cstdint>
#include <cstddef>

// ---------------------------------------------------------------------------
// Types for CDNA5 WMMA (gfx1250, wave32)
// ---------------------------------------------------------------------------
typedef __attribute__((ext_vector_type(16))) __bf16 v16bf;
typedef __attribute__((ext_vector_type(8)))  float  v8f;

// ---------------------------------------------------------------------------
// Device math helpers
// ---------------------------------------------------------------------------
__device__ __forceinline__ float dsilu(float x) { return x / (1.f + __expf(-x)); }
__device__ __forceinline__ float dgelu(float x) { return 0.5f * x * (1.f + erff(x * 0.70710678118654752f)); }
__device__ __forceinline__ float dsoftplus(float x) { return (x > 20.f) ? x : log1pf(__expf(x)); }

__device__ __forceinline__ float block_sum(float v, float* sm) {
  int lane = threadIdx.x & 31, wid = threadIdx.x >> 5;
#pragma unroll
  for (int o = 16; o > 0; o >>= 1) v += __shfl_down(v, o, 32);
  if (lane == 0) sm[wid] = v;
  __syncthreads();
  if (wid == 0) {
    int nw = (blockDim.x + 31) >> 5;
    float r = (lane < nw) ? sm[lane] : 0.f;
#pragma unroll
    for (int o = 16; o > 0; o >>= 1) r += __shfl_down(r, o, 32);
    if (lane == 0) sm[0] = r;
  }
  __syncthreads();
  float r = sm[0];
  __syncthreads();
  return r;
}

// ---------------------------------------------------------------------------
// WMMA GEMM helpers: fragment element i maps to K = (i>>3)*16 + khalf*8 + (i&7)
// i.e. two 8-float contiguous runs at k0+khalf*8 and k0+16+khalf*8 -> 4x b128.
// ---------------------------------------------------------------------------
__device__ __forceinline__ void cvt16(v16bf& dst, const float4& x0, const float4& x1,
                                      const float4& x2, const float4& x3) {
  dst[0]  = (__bf16)x0.x; dst[1]  = (__bf16)x0.y; dst[2]  = (__bf16)x0.z; dst[3]  = (__bf16)x0.w;
  dst[4]  = (__bf16)x1.x; dst[5]  = (__bf16)x1.y; dst[6]  = (__bf16)x1.z; dst[7]  = (__bf16)x1.w;
  dst[8]  = (__bf16)x2.x; dst[9]  = (__bf16)x2.y; dst[10] = (__bf16)x2.z; dst[11] = (__bf16)x2.w;
  dst[12] = (__bf16)x3.x; dst[13] = (__bf16)x3.y; dst[14] = (__bf16)x3.z; dst[15] = (__bf16)x3.w;
}

__device__ __forceinline__ v8f wmma_step_vec(const float* __restrict__ ap,
                                             const float* __restrict__ bp,
                                             int k0, int khalf, v8f acc) {
  const float4* a0 = (const float4*)(ap + k0 + khalf * 8);
  const float4* a1 = (const float4*)(ap + k0 + 16 + khalf * 8);
  const float4* b0 = (const float4*)(bp + k0 + khalf * 8);
  const float4* b1 = (const float4*)(bp + k0 + 16 + khalf * 8);
  float4 A0 = a0[0], A1 = a0[1], A2 = a1[0], A3 = a1[1];
  float4 B0 = b0[0], B1 = b0[1], B2 = b1[0], B3 = b1[1];
  v16bf a, b;
  cvt16(a, A0, A1, A2, A3);
  cvt16(b, B0, B1, B2, B3);
  return __builtin_amdgcn_wmma_f32_16x16x32_bf16(false, a, false, b, (short)0, acc,
                                                 false, false);
}

// ---------------------------------------------------------------------------
// Generic bf16 WMMA GEMM:  C[M,N] = act( A[M,K] * B[N,K]^T + bias[N] )  (+= C)
// 8 waves / block, each wave computes one 16x16 tile.
// Interior tiles: unguarded b128 fragment loads, K unrolled by 64 (2 WMMA/iter).
// Edge tiles / K tails: guarded scalar path (zero-padded; bf16 zero is exact).
// act: 0=none 1=silu 2=gelu(exact) 3=softplus
// ---------------------------------------------------------------------------
__global__ void gemm_bf16_wmma_k(const float* __restrict__ A, int lda,
                                 const float* __restrict__ B, int ldb,
                                 float* __restrict__ C, int ldc,
                                 int M, int N, int K,
                                 const float* __restrict__ bias,
                                 int accumulate, int act) {
  const int w    = threadIdx.x >> 5;
  const int lane = threadIdx.x & 31;
  const int tm = blockIdx.y * 2 + (w >> 2);
  const int tn = blockIdx.x * 4 + (w & 3);
  const int m0 = tm * 16, n0 = tn * 16;
  if (m0 >= M || n0 >= N) return;          // wave-uniform exit; EXEC stays all-1s
  const int row   = lane & 15;
  const int khalf = lane >> 4;
  const int am = m0 + row;
  const int bn = n0 + row;
  v8f acc = {0.f, 0.f, 0.f, 0.f, 0.f, 0.f, 0.f, 0.f};

  const bool interior = (m0 + 16 <= M) && (n0 + 16 <= N) && (((lda | ldb) & 3) == 0);
  int k0 = 0;
  if (interior) {
    const float* ap = A + (size_t)am * lda;
    const float* bp = B + (size_t)bn * ldb;
    for (; k0 + 64 <= K; k0 += 64) {
      acc = wmma_step_vec(ap, bp, k0,      khalf, acc);
      acc = wmma_step_vec(ap, bp, k0 + 32, khalf, acc);
    }
    for (; k0 + 32 <= K; k0 += 32)
      acc = wmma_step_vec(ap, bp, k0, khalf, acc);
  }
  for (; k0 < K; k0 += 32) {               // guarded tail / edge-tile path
    v16bf a, b;
#pragma unroll
    for (int i = 0; i < 16; ++i) {
      int kk = k0 + ((i >> 3) << 4) + (khalf << 3) + (i & 7);
      float av = (am < M && kk < K) ? A[(size_t)am * lda + kk] : 0.f;
      float bv = (bn < N && kk < K) ? B[(size_t)bn * ldb + kk] : 0.f;
      a[i] = (__bf16)av;
      b[i] = (__bf16)bv;
    }
    acc = __builtin_amdgcn_wmma_f32_16x16x32_bf16(false, a, false, b, (short)0, acc,
                                                  false, false);
  }

  // C/D layout: lane -> N = lane&15 ; VGPR r -> M = r + 8*(lane>>4)
  const int cn = n0 + (lane & 15);
  const int mb = m0 + ((lane >> 4) << 3);
  const float bv = (bias && cn < N) ? bias[cn] : 0.f;
#pragma unroll
  for (int r = 0; r < 8; ++r) {
    int cm = mb + r;
    if (cm < M && cn < N) {
      float v = acc[r] + bv;
      if (act == 1) v = dsilu(v);
      else if (act == 2) v = dgelu(v);
      else if (act == 3) v = dsoftplus(v);
      float* cp = C + (size_t)cm * ldc + cn;
      *cp = accumulate ? (*cp + v) : v;
    }
  }
}

// ---------------------------------------------------------------------------
// Small elementwise / shaping kernels
// ---------------------------------------------------------------------------
__global__ void c0_conv_k(const float* __restrict__ lat, const float* __restrict__ w,
                          const float* __restrict__ b, float* __restrict__ u) {
  int idx = blockIdx.x * blockDim.x + threadIdx.x;
  if (idx >= 32 * 512) return;
  int pos = idx % 512, o = idx / 512;
  float s = b[o];
  for (int i = 0; i < 18; ++i)
#pragma unroll
    for (int k = 0; k < 3; ++k) {
      int pp = pos + k - 1;
      if (pp >= 0 && pp < 512) s += lat[i * 512 + pp] * w[(o * 18 + i) * 3 + k];
    }
  u[o * 512 + pos] = s;   // u[l=o][d=pos]: (1,32ch,512) reinterpreted as (L=32,D=512)
}

__global__ void m2_dt_k(const float* __restrict__ zx, const float* __restrict__ dt_bias,
                        float* __restrict__ dtb) {
  int idx = blockIdx.x * blockDim.x + threadIdx.x;
  if (idx >= 32 * 16) return;
  int g = idx % 16, l = idx / 16;
  dtb[l * 16 + g] = dsoftplus(zx[l * 2320 + 2304 + g] + dt_bias[g]);
}

__global__ void m2_conv_k(const float* __restrict__ zx, const float* __restrict__ w,
                          const float* __restrict__ b, float* __restrict__ xc) {
  int idx = blockIdx.x * blockDim.x + threadIdx.x;
  if (idx >= 32 * 1280) return;
  int c = idx % 1280, l = idx / 1280;
  float s = b[c];
#pragma unroll
  for (int k = 0; k < 4; ++k) {
    int ll = l - 3 + k;                     // causal pad 3
    if (ll >= 0) s += zx[ll * 2320 + 1024 + c] * w[c * 4 + k];
  }
  xc[l * 1280 + c] = dsilu(s);
}

// mamba2 selective scan: 1024 threads = (g,p), state h[128] per thread, L=32 steps
__global__ void m2_scan_k(const float* __restrict__ xc, const float* __restrict__ dtb,
                          const float* __restrict__ A_log, const float* __restrict__ D,
                          float* __restrict__ y) {
  int tid = threadIdx.x;
  int g = tid >> 6, p = tid & 63;
  float A  = -__expf(A_log[g]);
  float Dg = D[g];
  float h[128];
  for (int n = 0; n < 128; ++n) h[n] = 0.f;
  for (int l = 0; l < 32; ++l) {
    float dt  = dtb[l * 16 + g];
    float x   = xc[l * 1280 + g * 64 + p];
    float dA  = __expf(dt * A);
    float dtx = dt * x;
    const float* Bp = xc + l * 1280 + 1024;
    const float* Cp = xc + l * 1280 + 1152;
    float acc = 0.f;
    for (int n = 0; n < 128; ++n) {
      h[n] = dA * h[n] + dtx * Bp[n];
      acc += h[n] * Cp[n];
    }
    y[l * 1024 + g * 64 + p] = acc + Dg * x;
  }
}

// gated RMSNorm: out = (y*silu(z)) * rsqrt(mean((y*silu(z))^2)+eps) * norm_w
__global__ void m2_rms_gate_k(const float* __restrict__ y, const float* __restrict__ zx,
                              const float* __restrict__ nw, float* __restrict__ out) {
  __shared__ float sm[32];
  int row = blockIdx.x;
  float ssum = 0.f;
  for (int i = threadIdx.x; i < 1024; i += blockDim.x) {
    float g = y[row * 1024 + i] * dsilu(zx[row * 2320 + i]);
    ssum += g * g;
  }
  float ms = block_sum(ssum, sm) / 1024.f;
  float r = rsqrtf(ms + 1e-5f);
  for (int i = threadIdx.x; i < 1024; i += blockDim.x) {
    float g = y[row * 1024 + i] * dsilu(zx[row * 2320 + i]);
    out[row * 1024 + i] = g * r * nw[i];
  }
}

__global__ void c1_im2col_k(const float* __restrict__ m2o, float* __restrict__ bc) {
  int idx = blockIdx.x * blockDim.x + threadIdx.x;
  if (idx >= 32 * 1536) return;
  int j = idx % 1536, t = idx / 1536;
  int i = j / 3, k = j % 3;
  int tt = t + k - 1;
  bc[idx] = (tt >= 0 && tt < 32) ? m2o[tt * 512 + i] : 0.f;
}

__global__ void transpose_k(const float* __restrict__ in, float* __restrict__ out,
                            int R, int Ccols) {
  int idx = blockIdx.x * blockDim.x + threadIdx.x;
  if (idx >= R * Ccols) return;
  int c = idx % Ccols, r = idx / Ccols;
  out[(size_t)c * R + r] = in[idx];
}

// LayerNorm over last dim, one block per row
__global__ void layernorm_k(const float* __restrict__ in, float* __restrict__ out, int C,
                            const float* __restrict__ w, const float* __restrict__ b) {
  __shared__ float sm[32];
  int row = blockIdx.x;
  const float* x = in + (size_t)row * C;
  float* y = out + (size_t)row * C;
  float s = 0.f;
  for (int i = threadIdx.x; i < C; i += blockDim.x) s += x[i];
  float mean = block_sum(s, sm) / (float)C;
  float v = 0.f;
  for (int i = threadIdx.x; i < C; i += blockDim.x) { float d = x[i] - mean; v += d * d; }
  float rstd = rsqrtf(block_sum(v, sm) / (float)C + 1e-5f);
  for (int i = threadIdx.x; i < C; i += blockDim.x)
    y[i] = (x[i] - mean) * rstd * w[i] + b[i];
}

__global__ void ew_act_k(float* __restrict__ x, long n, int act) {
  long idx = (long)blockIdx.x * blockDim.x + threadIdx.x;
  if (idx >= n) return;
  float v = x[idx];
  x[idx] = (act == 1) ? dsilu(v) : (act == 2) ? dgelu(v) : dsoftplus(v);
}

// depthwise 3x3 conv, HWC, input row stride ldin (to slice xin out of xz), pad 1
__global__ void dwconv3x3_k(const float* __restrict__ in, int ldin,
                            const float* __restrict__ w, const float* __restrict__ b,
                            float* __restrict__ out, int H, int W, int C, int act) {
  int idx = blockIdx.x * blockDim.x + threadIdx.x;
  if (idx >= H * W * C) return;
  int c = idx % C, l = idx / C;
  int h = l / W, x0 = l % W;
  float s = b[c];
#pragma unroll
  for (int ky = 0; ky < 3; ++ky)
#pragma unroll
    for (int kx = 0; kx < 3; ++kx) {
      int hh = h + ky - 1, ww = x0 + kx - 1;
      if (hh >= 0 && hh < H && ww >= 0 && ww < W)
        s += in[(size_t)(hh * W + ww) * ldin + c] * w[c * 9 + ky * 3 + kx];
    }
  if (act == 1) s = dsilu(s);
  out[(size_t)l * C + c] = s;
}

// build 4-direction sequences: 0=row-major, 1=col-major, 2=rev(0), 3=rev(1)
__global__ void gather_xs_k(const float* __restrict__ xin, float* __restrict__ xs,
                            int H, int W, int d) {
  int idx = blockIdx.x * blockDim.x + threadIdx.x;
  int L = H * W;
  if (idx >= L * d) return;
  int c = idx % d, l = idx / d;
  int h = l / W, w0 = l % W;
  int ltr = w0 * H + h;
  float v = xin[idx];
  xs[(size_t)(0 * L + l) * d + c]             = v;
  xs[(size_t)(1 * L + ltr) * d + c]           = v;
  xs[(size_t)(2 * L + (L - 1 - l)) * d + c]   = v;
  xs[(size_t)(3 * L + (L - 1 - ltr)) * d + c] = v;
}

// ss2d selective scan: thread = (k,c), state h[16] in registers, sequential over L.
// y (+D*x) written in-place over dts (same thread/element, read-before-write).
// Latency-bound critical path: prefetch 8 steps (8*4KB) ahead.
__global__ void ss2d_scan_k(const float* __restrict__ xs, float* __restrict__ dts,
                            const float* __restrict__ xdbl, int ldx, int roff,
                            const float* __restrict__ A_log, const float* __restrict__ D,
                            int d, int L) {
  int tid = blockIdx.x * blockDim.x + threadIdx.x;
  if (tid >= 4 * d) return;
  int k = tid / d, c = tid % d;
  float A[16], h[16];
#pragma unroll
  for (int n = 0; n < 16; ++n) {
    A[n] = -__expf(A_log[(size_t)(k * d + c) * 16 + n]);
    h[n] = 0.f;
  }
  float Dv = D[k * d + c];
  const float* xsk = xs + (size_t)k * L * d;
  float* dtk = dts + (size_t)k * L * d;
  const float* xd = xdbl + (size_t)k * L * ldx;
  for (int l = 0; l < L; ++l) {
    int lp = (l + 8 < L) ? (l + 8) : (L - 1);        // prefetch-ahead distance
    __builtin_prefetch(&dtk[(size_t)lp * d + c], 0, 1);
    __builtin_prefetch(&xsk[(size_t)lp * d + c], 0, 1);
    __builtin_prefetch(xd + (size_t)lp * ldx + roff, 0, 1);
    float dt  = dtk[(size_t)l * d + c];
    float x   = xsk[(size_t)l * d + c];
    float dtx = dt * x;
    const float* Bp = xd + (size_t)l * ldx + roff;
    const float* Cp = Bp + 16;
    float y = 0.f;
#pragma unroll
    for (int n = 0; n < 16; ++n) {
      h[n] = __expf(dt * A[n]) * h[n] + dtx * Bp[n];
      y += h[n] * Cp[n];
    }
    dtk[(size_t)l * d + c] = y + Dv * x;   // alias write after reads
  }
}

// combine 4 directions back to row-major positions
__global__ void ss2d_combine_k(const float* __restrict__ yd, float* __restrict__ out,
                               int H, int W, int d) {
  int idx = blockIdx.x * blockDim.x + threadIdx.x;
  int L = H * W;
  if (idx >= L * d) return;
  int c = idx % d, l = idx / d;
  int h = l / W, w0 = l % W;
  int ltr = w0 * H + h;
  float v = yd[(size_t)(0 * L + l) * d + c]
          + yd[(size_t)(2 * L + (L - 1 - l)) * d + c]
          + yd[(size_t)(1 * L + ltr) * d + c]
          + yd[(size_t)(3 * L + (L - 1 - ltr)) * d + c];
  out[idx] = v;
}

__global__ void gate_silu_k(float* __restrict__ y, const float* __restrict__ xz,
                            int ld, int off, int d, long total) {
  long idx = (long)blockIdx.x * blockDim.x + threadIdx.x;
  if (idx >= total) return;
  int c = (int)(idx % d);
  long l = idx / d;
  y[idx] *= dsilu(xz[(size_t)l * ld + off + c]);
}

__global__ void im2col3x3_k(const float* __restrict__ in, float* __restrict__ col,
                            int H, int W, int Cin) {
  long idx = (long)blockIdx.x * blockDim.x + threadIdx.x;
  long total = (long)H * W * Cin * 9;
  if (idx >= total) return;
  int j = (int)(idx % 9);
  long t = idx / 9;
  int c = (int)(t % Cin);
  int l = (int)(t / Cin);
  int h = l / W, w0 = l % W;
  int ky = j / 3, kx = j % 3;
  int hh = h + ky - 1, ww = w0 + kx - 1;
  float v = 0.f;
  if (hh >= 0 && hh < H && ww >= 0 && ww < W) v = in[(size_t)(hh * W + ww) * Cin + c];
  col[(size_t)l * (Cin * 9) + c * 9 + j] = v;   // patch layout matches OIHW weight rows
}

__global__ void pixshuf_k(const float* __restrict__ in, float* __restrict__ out) {
  int idx = blockIdx.x * blockDim.x + threadIdx.x;
  if (idx >= 4096 * 128) return;
  int c = idx % 128, l = idx / 128;
  int h = l / 64, w0 = l % 64;
  out[idx] = in[(size_t)((h >> 1) * 32 + (w0 >> 1)) * 512 + (c * 4 + (h & 1) * 2 + (w0 & 1))];
}

__global__ void avgpool3_k(const float* __restrict__ in, float* __restrict__ out,
                           int H, int W, int C) {
  int idx = blockIdx.x * blockDim.x + threadIdx.x;
  if (idx >= H * W * C) return;
  int c = idx % C, l = idx / C;
  int h = l / W, w0 = l % W;
  float s = 0.f;
#pragma unroll
  for (int dy = -1; dy <= 1; ++dy)
#pragma unroll
    for (int dx = -1; dx <= 1; ++dx) {
      int hh = h + dy, ww = w0 + dx;
      if (hh >= 0 && hh < H && ww >= 0 && ww < W) s += in[(size_t)(hh * W + ww) * C + c];
    }
  out[idx] = s / 9.f;
}

// ---------------------------------------------------------------------------
// Host-side orchestration
// ---------------------------------------------------------------------------
namespace {

struct WS {
  float* base;
  size_t off;
  float* alloc(size_t n) {
    float* p = base + off;
    off += (n + 63) & ~(size_t)63;
    return p;
  }
};

struct SSP {
  const float *ln1_w, *ln1_b, *in_proj_w, *conv_w, *conv_b, *x_proj_w, *dt_w, *dt_b,
              *A_log, *D, *ln_w, *ln_b, *out_proj_w;
};

inline dim3 g1(long n, int b = 256) { return dim3((unsigned)((n + b - 1) / b)); }

void launch_gemm(const float* A, int lda, const float* B, int ldb, float* C, int ldc,
                 int M, int N, int K, const float* bias, int acc, int act,
                 hipStream_t s) {
  dim3 blk(256);
  dim3 grd((N + 63) / 64, (M + 31) / 32);
  gemm_bf16_wmma_k<<<grd, blk, 0, s>>>(A, lda, B, ldb, C, ldc, M, N, K, bias, acc, act);
}

// x: [L][C] input (residual source). out: [L][C], gets x + ss2d(ln1(x)).
// WS passed BY VALUE so scratch is reused across the two VSS blocks.
void run_vss(const float* x, float* out, const SSP& p, int C, int H, int W, WS t,
             hipStream_t s) {
  const int d = 2 * C, L = H * W, r = C / 16, cdim = r + 32;
  hipMemcpyAsync(out, x, (size_t)L * C * sizeof(float), hipMemcpyDeviceToDevice, s);

  float* hln = t.alloc((size_t)L * C);
  layernorm_k<<<L, 256, 0, s>>>(x, hln, C, p.ln1_w, p.ln1_b);

  float* xz = t.alloc((size_t)L * 2 * d);
  launch_gemm(hln, C, p.in_proj_w, C, xz, 2 * d, L, 2 * d, C, nullptr, 0, 0, s);

  float* xin = t.alloc((size_t)L * d);
  dwconv3x3_k<<<g1((long)L * d), 256, 0, s>>>(xz, 2 * d, p.conv_w, p.conv_b, xin, H, W, d, 1);

  float* xs = t.alloc((size_t)4 * L * d);
  gather_xs_k<<<g1((long)L * d), 256, 0, s>>>(xin, xs, H, W, d);

  float* xdbl = t.alloc((size_t)4 * L * cdim);
  for (int k = 0; k < 4; ++k)
    launch_gemm(xs + (size_t)k * L * d, d, p.x_proj_w + (size_t)k * cdim * d, d,
                xdbl + (size_t)k * L * cdim, cdim, L, cdim, d, nullptr, 0, 0, s);

  float* dts = t.alloc((size_t)4 * L * d);
  for (int k = 0; k < 4; ++k)
    launch_gemm(xdbl + (size_t)k * L * cdim, cdim, p.dt_w + (size_t)k * d * r, r,
                dts + (size_t)k * L * d, d, L, d, r, p.dt_b + (size_t)k * d, 0, 3, s);

  ss2d_scan_k<<<g1(4 * d), 256, 0, s>>>(xs, dts, xdbl, cdim, r, p.A_log, p.D, d, L);

  float* ycomb = t.alloc((size_t)L * d);
  ss2d_combine_k<<<g1((long)L * d), 256, 0, s>>>(dts, ycomb, H, W, d);

  float* yln = t.alloc((size_t)L * d);
  layernorm_k<<<L, 256, 0, s>>>(ycomb, yln, d, p.ln_w, p.ln_b);

  gate_silu_k<<<g1((long)L * d), 256, 0, s>>>(yln, xz, 2 * d, d, d, (long)L * d);

  // out += yln * out_proj_w^T   (residual already in out)
  launch_gemm(yln, d, p.out_proj_w, d, out, C, L, C, d, nullptr, 1, 0, s);
}

}  // namespace

extern "C" void kernel_launch(void* const* d_in, const int* in_sizes, int n_in,
                              void* d_out, int out_size, void* d_ws, size_t ws_size,
                              hipStream_t stream) {
  (void)in_sizes; (void)n_in; (void)out_size; (void)ws_size;
  auto in = [&](int i) { return (const float*)d_in[i]; };

  // setup_inputs() dict order (nested dicts depth-first in insertion order)
  const float* lat        = in(0);
  const float* c0_w       = in(1);
  const float* c0_b       = in(2);
  const float* m2_in_w    = in(3);
  const float* m2_conv_w  = in(4);
  const float* m2_conv_b  = in(5);
  const float* m2_dt_bias = in(6);
  const float* m2_A_log   = in(7);
  const float* m2_D       = in(8);
  const float* m2_norm_w  = in(9);
  const float* m2_out_w   = in(10);
  const float* c1_w       = in(11);
  const float* c1_b       = in(12);
  SSP ss1 = { in(13), in(14), in(15), in(16), in(17), in(18), in(19), in(20),
              in(21), in(22), in(23), in(24), in(25) };
  SSP ss2 = { in(26), in(27), in(28), in(29), in(30), in(31), in(32), in(33),
              in(34), in(35), in(36), in(37), in(38) };
  const float* pw1_w  = in(39); const float* pw1_b  = in(40);
  const float* ln64_w = in(41); const float* ln64_b = in(42);
  const float* dw2_w  = in(43); const float* dw2_b  = in(44);
  const float* pw2_w  = in(45); const float* pw2_b  = in(46);
  const float* ln128_w = in(47); const float* ln128_b = in(48);
  const float* up_w   = in(49); const float* up_b   = in(50);
  const float* cv1_w  = in(51); const float* cv1_b  = in(52);
  const float* cv2_w  = in(53); const float* cv2_b  = in(54);
  const float* fc_w   = in(55); const float* fc_b   = in(56);

  WS ws{ (float*)d_ws, 0 };
  // persistent buffers
  float* x1      = ws.alloc(1024 * 32);     // VSS1 input  [L=1024][C=32]
  float* vss1out = ws.alloc(1024 * 32);     // VSS1 output
  float* fco     = ws.alloc(4096 * 512);    // fc conv out (HWC) = VSS2 input
  float* out2    = ws.alloc(4096 * 512);    // VSS2 output

  // ---- Stage 1: c0 conv1d + mamba2 + c1 conv1d (scoped scratch) ----
  {
    WS t = ws;
    float* u = t.alloc(32 * 512);
    c0_conv_k<<<g1(32 * 512), 256, 0, stream>>>(lat, c0_w, c0_b, u);

    float* zx = t.alloc(32 * 2320);
    launch_gemm(u, 512, m2_in_w, 512, zx, 2320, 32, 2320, 512, nullptr, 0, 0, stream);

    float* dtb = t.alloc(32 * 16);
    m2_dt_k<<<g1(32 * 16), 256, 0, stream>>>(zx, m2_dt_bias, dtb);

    float* xc = t.alloc(32 * 1280);
    m2_conv_k<<<g1(32 * 1280), 256, 0, stream>>>(zx, m2_conv_w, m2_conv_b, xc);

    float* ym = t.alloc(32 * 1024);
    m2_scan_k<<<1, 1024, 0, stream>>>(xc, dtb, m2_A_log, m2_D, ym);

    float* yn = t.alloc(32 * 1024);
    m2_rms_gate_k<<<32, 256, 0, stream>>>(ym, zx, m2_norm_w, yn);

    float* m2o = t.alloc(32 * 512);
    launch_gemm(yn, 1024, m2_out_w, 1024, m2o, 512, 32, 512, 1024, nullptr, 0, 0, stream);

    float* bc1 = t.alloc(32 * 1536);
    c1_im2col_k<<<g1(32 * 1536), 256, 0, stream>>>(m2o, bc1);

    float* c1o = t.alloc(32 * 1024);   // [t=32][o=1024]
    launch_gemm(bc1, 1536, c1_w, 1536, c1o, 1024, 32, 1024, 1536, c1_b, 0, 0, stream);

    // (1,1024,32) -> transpose(1,0,2): conv channels become sequence positions
    transpose_k<<<g1(32 * 1024), 256, 0, stream>>>(c1o, x1, 32, 1024);
  }

  // ---- Stage 2: VSS block 1 (C=32, 32x32) ----
  run_vss(x1, vss1out, ss1, 32, 32, 32, ws, stream);

  // ---- Stage 3: conv chain to 64x64x512 (scoped scratch) ----
  {
    WS t = ws;
    float* t1 = t.alloc(1024 * 64);
    launch_gemm(vss1out, 32, pw1_w, 32, t1, 64, 1024, 64, 32, pw1_b, 0, 0, stream);
    float* t1b = t.alloc(1024 * 64);
    layernorm_k<<<1024, 256, 0, stream>>>(t1, t1b, 64, ln64_w, ln64_b);
    ew_act_k<<<g1(1024 * 64), 256, 0, stream>>>(t1b, 1024 * 64, 2);

    float* t2 = t.alloc(1024 * 64);
    dwconv3x3_k<<<g1(1024 * 64), 256, 0, stream>>>(t1b, 64, dw2_w, dw2_b, t2, 32, 32, 64, 0);

    float* t3 = t.alloc(1024 * 128);
    launch_gemm(t2, 64, pw2_w, 64, t3, 128, 1024, 128, 64, pw2_b, 0, 0, stream);
    float* t3b = t.alloc(1024 * 128);
    layernorm_k<<<1024, 256, 0, stream>>>(t3, t3b, 128, ln128_w, ln128_b);
    ew_act_k<<<g1(1024 * 128), 256, 0, stream>>>(t3b, 1024 * 128, 2);

    float* col = t.alloc((size_t)4096 * 2304);   // shared im2col buffer (max size)

    // up: 3x3 conv 128->512 on 32x32
    im2col3x3_k<<<g1((long)1024 * 128 * 9), 256, 0, stream>>>(t3b, col, 32, 32, 128);
    float* upo = t.alloc(1024 * 512);
    launch_gemm(col, 1152, up_w, 1152, upo, 512, 1024, 512, 1152, up_b, 0, 0, stream);

    float* psb = t.alloc(4096 * 128);
    pixshuf_k<<<g1(4096 * 128), 256, 0, stream>>>(upo, psb);
    float* apb = t.alloc(4096 * 128);
    avgpool3_k<<<g1(4096 * 128), 256, 0, stream>>>(psb, apb, 64, 64, 128);

    // cv1: 3x3 128->128 on 64x64
    im2col3x3_k<<<g1((long)4096 * 128 * 9), 256, 0, stream>>>(apb, col, 64, 64, 128);
    float* cv1o = t.alloc(4096 * 128);
    launch_gemm(col, 1152, cv1_w, 1152, cv1o, 128, 4096, 128, 1152, cv1_b, 0, 0, stream);

    // cv2: 1x1 128->256
    float* cv2o = t.alloc(4096 * 256);
    launch_gemm(cv1o, 128, cv2_w, 128, cv2o, 256, 4096, 256, 128, cv2_b, 0, 0, stream);

    // fc: 3x3 256->512
    im2col3x3_k<<<g1((long)4096 * 256 * 9), 256, 0, stream>>>(cv2o, col, 64, 64, 256);
    launch_gemm(col, 2304, fc_w, 2304, fco, 512, 4096, 512, 2304, fc_b, 0, 0, stream);
  }

  // ---- Stage 4: VSS block 2 (C=512, 64x64) ----
  run_vss(fco, out2, ss2, 512, 64, 64, ws, stream);

  // ---- Stage 5: (L,B,C) -> (B,C,64,64): out[c][l] = out2[l][c] ----
  transpose_k<<<g1((long)4096 * 512), 256, 0, stream>>>(out2, (float*)d_out, 4096, 512);
}
